// CrossAttention_19224273616911
// MI455X (gfx1250) — matrix operand order repaired and checked by reading, not compiled, not run on hardware
//
#include <hip/hip_runtime.h>
#include <hip/hip_bf16.h>

// Problem constants (fixed by the reference)
#define BATCH 4
#define SEQ   2048          // Sx == Sy
#define DIM   1024
#define MTOT  (BATCH * SEQ) // 8192 flat rows

// Register blocking: each wave computes (2*16) x (4*16) = 32 x 64 of output
#define MT 2
#define NT 4

typedef __attribute__((ext_vector_type(16))) __bf16       v16bf;
typedef __attribute__((ext_vector_type(8)))  float        v8f;
typedef __attribute__((ext_vector_type(4)))  unsigned int u32x4;

union FragU { u32x4 u[2]; v16bf v; };

// float -> bf16, round-to-nearest-even (no reliance on cast lowering)
__device__ __forceinline__ __bf16 f2bf(float x) {
    union { float f; unsigned u; } a; a.f = x;
    unsigned r = a.u + 0x7FFFu + ((a.u >> 16) & 1u);
    unsigned short h = (unsigned short)(r >> 16);
    return __builtin_bit_cast(__bf16, h);
}

// A-matrix fragment (16x32 bf16, M x K). Lane m = lane&15.
// lane<16 : K = k0+[0..7], k0+[16..23] ; lane>=16 : K = k0+[8..15], k0+[24..31]
__device__ __forceinline__ v16bf load_frag_a(const __bf16* row, int k0, int lane) {
    const int off = k0 + ((lane & 16) ? 8 : 0);
    FragU f;
    f.u[0] = *reinterpret_cast<const u32x4*>(row + off);
    f.u[1] = *reinterpret_cast<const u32x4*>(row + off + 16);
    return f.v;
}

// B-matrix fragment (32x16 bf16, K x N), sourced from a row-major array whose
// ROW n is column n of B. Lane n = lane&15.
// lane<16 : K = k0+[0..7], k0+[16..23] ; lane>=16 : K = k0+[8..15], k0+[24..31]
__device__ __forceinline__ v16bf load_frag_b(const __bf16* row, int k0, int lane) {
    const int off = k0 + ((lane & 16) ? 8 : 0);
    FragU f;
    f.u[0] = *reinterpret_cast<const u32x4*>(row + off);
    f.u[1] = *reinterpret_cast<const u32x4*>(row + off + 16);
    return f.v;
}

__device__ __forceinline__ v8f wmma_bf16(v16bf a, v16bf b, v8f c) {
    return __builtin_amdgcn_wmma_f32_16x16x32_bf16(false, a, false, b, (short)0, c, false, false);
}

// ---------------- Pass 0: conversions ----------------
__global__ __launch_bounds__(256) void f32_to_bf16_kernel(const float* __restrict__ in,
                                                          __bf16* __restrict__ out, int n) {
    int i = blockIdx.x * 256 + threadIdx.x;
    if (i < n) out[i] = f2bf(in[i]);
}

// Wt[n][k] = W[k][n]  (bf16), so B-fragment loads are contiguous
__global__ __launch_bounds__(256) void transpose_to_bf16_kernel(const float* __restrict__ W,
                                                                __bf16* __restrict__ Wt) {
    int i = blockIdx.x * 256 + threadIdx.x; // i = n*DIM + k
    if (i < DIM * DIM) {
        int n = i >> 10;            // / DIM
        int k = i & (DIM - 1);
        Wt[i] = f2bf(W[(size_t)k * DIM + n]);
    }
}

// ---------------- Pass 1: QKV projection GEMM (32x64 per wave) ----------------
// out[row][n] = (sum_k A[row][k] * Wt[n][k] + bias[n]) * scale.
// transposed==0: row-major bf16 [MTOT][DIM]; transposed==1: [BATCH][DIM][SEQ]
__global__ __launch_bounds__(256) void qkv_gemm_kernel(const __bf16* __restrict__ A,
                                                       const __bf16* __restrict__ Wt,
                                                       const float*  __restrict__ bias,
                                                       __bf16* __restrict__ outp,
                                                       float scale, int transposed) {
    const int lane = threadIdx.x & 31;
    const int wave = threadIdx.x >> 5;
    const int tile = blockIdx.x * 8 + wave;
    const int ct = tile & 15;       // DIM/64 = 16 column macro-tiles
    const int rt = tile >> 4;       // MTOT/32 = 256 row macro-tiles

    const __bf16* arow[MT];
    const __bf16* brow[NT];
#pragma unroll
    for (int mi = 0; mi < MT; ++mi)
        arow[mi] = A + (size_t)(rt * 32 + mi * 16 + (lane & 15)) * DIM;
#pragma unroll
    for (int ni = 0; ni < NT; ++ni)
        brow[ni] = Wt + (size_t)(ct * 64 + ni * 16 + (lane & 15)) * DIM;

    v8f acc[MT][NT];
#pragma unroll
    for (int mi = 0; mi < MT; ++mi)
#pragma unroll
        for (int ni = 0; ni < NT; ++ni)
            acc[mi][ni] = (v8f){0.f, 0.f, 0.f, 0.f, 0.f, 0.f, 0.f, 0.f};

    for (int k0 = 0; k0 < DIM; k0 += 32) {
        v16bf a[MT], b[NT];
#pragma unroll
        for (int mi = 0; mi < MT; ++mi) a[mi] = load_frag_a(arow[mi], k0, lane);
#pragma unroll
        for (int ni = 0; ni < NT; ++ni) b[ni] = load_frag_b(brow[ni], k0, lane);
#pragma unroll
        for (int mi = 0; mi < MT; ++mi)
#pragma unroll
            for (int ni = 0; ni < NT; ++ni)
                acc[mi][ni] = wmma_bf16(a[mi], b[ni], acc[mi][ni]);
    }

    const int mb = (lane & 16) ? 8 : 0;
#pragma unroll
    for (int ni = 0; ni < NT; ++ni) {
        const int n = ct * 64 + ni * 16 + (lane & 15);
        const float bv = bias[n];
#pragma unroll
        for (int mi = 0; mi < MT; ++mi) {
            if (!transposed) {
#pragma unroll
                for (int i = 0; i < 8; ++i)
                    outp[(size_t)(rt * 32 + mi * 16 + mb + i) * DIM + n] =
                        f2bf((acc[mi][ni][i] + bv) * scale);
            } else {
#pragma unroll
                for (int i = 0; i < 8; ++i) {
                    int rowflat = rt * 32 + mi * 16 + mb + i;
                    int b_ = rowflat >> 11;        // / SEQ
                    int t  = rowflat & (SEQ - 1);
                    outp[((size_t)b_ * DIM + n) * SEQ + t] =
                        f2bf((acc[mi][ni][i] + bv) * scale);
                }
            }
        }
    }
}

// ---------------- Pass 2a: scores S = Q * K^T (32x64 per wave) ----------------
// B = K^T, so B's column t is K's row t (row-major K feeds B-fragments directly)
__global__ __launch_bounds__(256) void score_gemm_kernel(const __bf16* __restrict__ Q,
                                                         const __bf16* __restrict__ K,
                                                         float* __restrict__ S) {
    const int lane = threadIdx.x & 31;
    const int wave = threadIdx.x >> 5;
    const int tile = blockIdx.x * 8 + wave;
    const int ct = tile & 31;       // SEQ/64 = 32 column macro-tiles
    const int rt = tile >> 5;       // MTOT/32 = 256 row macro-tiles
    const int b_ = (rt * 32) >> 11; // batch of this row block

    const __bf16* arow[MT];
    const __bf16* brow[NT];
#pragma unroll
    for (int mi = 0; mi < MT; ++mi)
        arow[mi] = Q + (size_t)(rt * 32 + mi * 16 + (lane & 15)) * DIM;
#pragma unroll
    for (int ni = 0; ni < NT; ++ni)
        brow[ni] = K + ((size_t)b_ * SEQ + ct * 64 + ni * 16 + (lane & 15)) * DIM;

    v8f acc[MT][NT];
#pragma unroll
    for (int mi = 0; mi < MT; ++mi)
#pragma unroll
        for (int ni = 0; ni < NT; ++ni)
            acc[mi][ni] = (v8f){0.f, 0.f, 0.f, 0.f, 0.f, 0.f, 0.f, 0.f};

    for (int k0 = 0; k0 < DIM; k0 += 32) {
        v16bf a[MT], b[NT];
#pragma unroll
        for (int mi = 0; mi < MT; ++mi) a[mi] = load_frag_a(arow[mi], k0, lane);
#pragma unroll
        for (int ni = 0; ni < NT; ++ni) b[ni] = load_frag_b(brow[ni], k0, lane);
#pragma unroll
        for (int mi = 0; mi < MT; ++mi)
#pragma unroll
            for (int ni = 0; ni < NT; ++ni)
                acc[mi][ni] = wmma_bf16(a[mi], b[ni], acc[mi][ni]);
    }

    const int mb = (lane & 16) ? 8 : 0;
#pragma unroll
    for (int ni = 0; ni < NT; ++ni) {
        const int t = ct * 64 + ni * 16 + (lane & 15);
#pragma unroll
        for (int mi = 0; mi < MT; ++mi)
#pragma unroll
            for (int i = 0; i < 8; ++i)
                S[(size_t)(rt * 32 + mi * 16 + mb + i) * SEQ + t] = acc[mi][ni][i];
    }
}

// ---------------- Pass 2b: row softmax, f32 -> bf16 probs ----------------
__global__ __launch_bounds__(256) void softmax_kernel(const float* __restrict__ S,
                                                      __bf16* __restrict__ P) {
    const int row = blockIdx.x;
    const int tid = threadIdx.x;
    const float* s = S + (size_t)row * SEQ;
    __bf16* p = P + (size_t)row * SEQ;
    __shared__ float red[256];

    float mx = -3.4e38f;
    for (int t = tid; t < SEQ; t += 256) mx = fmaxf(mx, s[t]);
    red[tid] = mx; __syncthreads();
    for (int st = 128; st; st >>= 1) {
        if (tid < st) red[tid] = fmaxf(red[tid], red[tid + st]);
        __syncthreads();
    }
    mx = red[0];
    __syncthreads();

    float sum = 0.f;
    for (int t = tid; t < SEQ; t += 256) sum += __expf(s[t] - mx);
    red[tid] = sum; __syncthreads();
    for (int st = 128; st; st >>= 1) {
        if (tid < st) red[tid] += red[tid + st];
        __syncthreads();
    }
    const float inv = 1.0f / red[0];

    for (int t = tid; t < SEQ; t += 256) p[t] = f2bf(__expf(s[t] - mx) * inv);
}

// ---------------- Pass 2c: out = P * V (V stored transposed [B][DIM][SEQ]) ----------------
__global__ __launch_bounds__(256) void pv_gemm_kernel(const __bf16* __restrict__ P,
                                                      const __bf16* __restrict__ Vt,
                                                      float* __restrict__ out) {
    const int lane = threadIdx.x & 31;
    const int wave = threadIdx.x >> 5;
    const int tile = blockIdx.x * 8 + wave;
    const int ct = tile & 15;       // DIM/64 = 16 column macro-tiles
    const int rt = tile >> 4;       // MTOT/32 = 256 row macro-tiles
    const int b_ = (rt * 32) >> 11;

    const __bf16* arow[MT];
    const __bf16* brow[NT];
#pragma unroll
    for (int mi = 0; mi < MT; ++mi)
        arow[mi] = P + (size_t)(rt * 32 + mi * 16 + (lane & 15)) * SEQ;
#pragma unroll
    for (int ni = 0; ni < NT; ++ni)
        brow[ni] = Vt + ((size_t)b_ * DIM + ct * 64 + ni * 16 + (lane & 15)) * SEQ;

    v8f acc[MT][NT];
#pragma unroll
    for (int mi = 0; mi < MT; ++mi)
#pragma unroll
        for (int ni = 0; ni < NT; ++ni)
            acc[mi][ni] = (v8f){0.f, 0.f, 0.f, 0.f, 0.f, 0.f, 0.f, 0.f};

    for (int k0 = 0; k0 < SEQ; k0 += 32) {
        v16bf a[MT], b[NT];
#pragma unroll
        for (int mi = 0; mi < MT; ++mi) a[mi] = load_frag_a(arow[mi], k0, lane);
#pragma unroll
        for (int ni = 0; ni < NT; ++ni) b[ni] = load_frag_b(brow[ni], k0, lane);
#pragma unroll
        for (int mi = 0; mi < MT; ++mi)
#pragma unroll
            for (int ni = 0; ni < NT; ++ni)
                acc[mi][ni] = wmma_bf16(a[mi], b[ni], acc[mi][ni]);
    }

    const int mb = (lane & 16) ? 8 : 0;
#pragma unroll
    for (int ni = 0; ni < NT; ++ni) {
        const int d = ct * 64 + ni * 16 + (lane & 15);
#pragma unroll
        for (int mi = 0; mi < MT; ++mi)
#pragma unroll
            for (int i = 0; i < 8; ++i)
                out[(size_t)(rt * 32 + mi * 16 + mb + i) * DIM + d] = acc[mi][ni][i];
    }
}

extern "C" void kernel_launch(void* const* d_in, const int* in_sizes, int n_in,
                              void* d_out, int out_size, void* d_ws, size_t ws_size,
                              hipStream_t stream) {
    const float* x  = (const float*)d_in[0];
    const float* y  = (const float*)d_in[1];
    const float* Wq = (const float*)d_in[2];
    const float* bq = (const float*)d_in[3];
    const float* Wk = (const float*)d_in[4];
    const float* bk = (const float*)d_in[5];
    const float* Wv = (const float*)d_in[6];
    const float* bv = (const float*)d_in[7];
    float* out = (float*)d_out;

    char* ws = (char*)d_ws;
    const size_t MB = 1024u * 1024u;
    // Region A (dead after pass 1): xb, yb, W^T — reused for P in pass 2b.
    __bf16* xb  = (__bf16*)(ws + 0);        // 16 MB
    __bf16* yb  = (__bf16*)(ws + 16 * MB);  // 16 MB
    __bf16* Wqt = (__bf16*)(ws + 32 * MB);  //  2 MB
    __bf16* Wkt = (__bf16*)(ws + 34 * MB);  //  2 MB
    __bf16* Wvt = (__bf16*)(ws + 36 * MB);  //  2 MB
    __bf16* Pb  = (__bf16*)(ws + 0);        // 32 MB (aliases xb/yb, safe: written after pass 1)
    __bf16* Qb  = (__bf16*)(ws + 38 * MB);  // 16 MB
    __bf16* Kb  = (__bf16*)(ws + 54 * MB);  // 16 MB
    __bf16* Vtb = (__bf16*)(ws + 70 * MB);  // 16 MB [BATCH][DIM][SEQ]
    float*  S   = (float*) (ws + 86 * MB);  // 64 MB [MTOT][SEQ]

    const int nXY = MTOT * DIM; // 8,388,608 elements each for x and y
    const float qscale = 0.03125f; // 1/sqrt(1024)

    // Pass 0: convert inputs
    f32_to_bf16_kernel<<<nXY / 256, 256, 0, stream>>>(x, xb, nXY);
    f32_to_bf16_kernel<<<nXY / 256, 256, 0, stream>>>(y, yb, nXY);
    transpose_to_bf16_kernel<<<(DIM * DIM) / 256, 256, 0, stream>>>(Wq, Wqt);
    transpose_to_bf16_kernel<<<(DIM * DIM) / 256, 256, 0, stream>>>(Wk, Wkt);
    transpose_to_bf16_kernel<<<(DIM * DIM) / 256, 256, 0, stream>>>(Wv, Wvt);

    // Pass 1: projections. Wave-tiles = (MTOT/32)*(DIM/64) = 4096; 8 waves/block.
    const int qkvBlocks = (MTOT / 32) * (DIM / 64) / 8; // 512
    qkv_gemm_kernel<<<qkvBlocks, 256, 0, stream>>>(xb, Wqt, bq, Qb, qscale, 0);
    qkv_gemm_kernel<<<qkvBlocks, 256, 0, stream>>>(yb, Wkt, bk, Kb, 1.0f, 0);
    qkv_gemm_kernel<<<qkvBlocks, 256, 0, stream>>>(yb, Wvt, bv, Vtb, 1.0f, 1);

    // Pass 2a: scores. Wave-tiles = (MTOT/32)*(SEQ/64) = 8192; 8 waves/block.
    const int scoreBlocks = (MTOT / 32) * (SEQ / 64) / 8; // 1024
    score_gemm_kernel<<<scoreBlocks, 256, 0, stream>>>(Qb, Kb, S);

    // Pass 2b: softmax over each of the MTOT rows
    softmax_kernel<<<MTOT, 256, 0, stream>>>(S, Pb);

    // Pass 2c: out = P * V
    const int pvBlocks = (MTOT / 32) * (DIM / 64) / 8; // 512
    pv_gemm_kernel<<<pvBlocks, 256, 0, stream>>>(Pb, Vtb, out);
}
